// NTXentLoss_26371099197648
// MI455X (gfx1250) — matrix operand order repaired and checked by reading, not compiled, not run on hardware
//
#include <hip/hip_runtime.h>
#include <hip/hip_bf16.h>
#include <math.h>
#include <stdint.h>

typedef __attribute__((ext_vector_type(16))) _Float16 v16h;
typedef __attribute__((ext_vector_type(8)))  _Float16 v8h;
typedef __attribute__((ext_vector_type(8)))  float    v8f;

#define D_DIM     256
#define NCHUNKS   8
#define EPS       1e-8f

// LDS staging geometry for the sim kernel.
#define TILE_N    32                    // columns staged per buffer
#define COL_BYTES 528                   // 512 B of K data + 16 B pad (bank-optimal)
#define BUF_BYTES (TILE_N * COL_BYTES)  // 16896 B per buffer

// ---------------------------------------------------------------------------
// CDNA5 async global->LDS copy (16 B per lane), tracked by ASYNCcnt.
// VDST = LDS byte offset in a VGPR, VADDR = 64-bit global address (GV mode).
// ---------------------------------------------------------------------------
static __device__ __forceinline__ void async_copy_b128(unsigned lds_off,
                                                       const void* gptr) {
  asm volatile("global_load_async_to_lds_b128 %0, %1, off"
               :: "v"(lds_off), "v"(gptr)
               : "memory");
}
static __device__ __forceinline__ void wait_async_all() {
  asm volatile("s_wait_asynccnt 0" ::: "memory");
}

// ---------------------------------------------------------------------------
// Kernel 1: normalize rows of [z_i; z_j] into an f16 copy (one wave per row).
// ---------------------------------------------------------------------------
__global__ void nt_normalize(const float* __restrict__ zi,
                             const float* __restrict__ zj,
                             _Float16* __restrict__ zh, int n) {
  const int wave = threadIdx.x >> 5;
  const int lane = threadIdx.x & 31;
  const int row  = blockIdx.x * (blockDim.x >> 5) + wave;

  const float* src = (row < n) ? (zi + (size_t)row * D_DIM)
                               : (zj + (size_t)(row - n) * D_DIM);
  float x[8];
  const float4* p = (const float4*)(src + lane * 8);
  float4 f0 = p[0], f1 = p[1];
  x[0]=f0.x; x[1]=f0.y; x[2]=f0.z; x[3]=f0.w;
  x[4]=f1.x; x[5]=f1.y; x[6]=f1.z; x[7]=f1.w;

  float s = 0.f;
#pragma unroll
  for (int i = 0; i < 8; ++i) s += x[i] * x[i];
  s += __shfl_xor(s, 1);  s += __shfl_xor(s, 2);
  s += __shfl_xor(s, 4);  s += __shfl_xor(s, 8);
  s += __shfl_xor(s, 16);

  const float scale = 1.f / fmaxf(sqrtf(s), EPS);

  v8h h;
#pragma unroll
  for (int i = 0; i < 8; ++i) h[i] = (_Float16)(x[i] * scale);
  *(v8h*)(zh + (size_t)row * D_DIM + lane * 8) = h;
}

// ---------------------------------------------------------------------------
// Kernel 2: positives in full fp32 from the original inputs (one wave per j).
// ---------------------------------------------------------------------------
__global__ void nt_positives(const float* __restrict__ zi,
                             const float* __restrict__ zj,
                             float* __restrict__ pos, int n) {
  const int wave = threadIdx.x >> 5;
  const int lane = threadIdx.x & 31;
  const int j    = blockIdx.x * (blockDim.x >> 5) + wave;
  if (j >= n) return;

  const float4* pa = (const float4*)(zi + (size_t)j * D_DIM + lane * 8);
  const float4* pb = (const float4*)(zj + (size_t)j * D_DIM + lane * 8);
  float d = 0.f, na = 0.f, nb = 0.f;
#pragma unroll
  for (int q = 0; q < 2; ++q) {
    float4 a = pa[q], b = pb[q];
    d  += a.x*b.x + a.y*b.y + a.z*b.z + a.w*b.w;
    na += a.x*a.x + a.y*a.y + a.z*a.z + a.w*a.w;
    nb += b.x*b.x + b.y*b.y + b.z*b.z + b.w*b.w;
  }
#pragma unroll
  for (int m = 1; m < 32; m <<= 1) {
    d  += __shfl_xor(d,  m);
    na += __shfl_xor(na, m);
    nb += __shfl_xor(nb, m);
  }
  if (lane == 0)
    pos[j] = d / (fmaxf(sqrtf(na), EPS) * fmaxf(sqrtf(nb), EPS));
}

// ---------------------------------------------------------------------------
// Kernel 3: fused sim-GEMM + exp + row-sum.
// Block = 4 waves x 16-row strips (64 rows) over one column chunk. The block
// double-buffers 32-column B tiles in LDS via async global->LDS DMA; each
// buffer feeds 16 v_wmma_f32_16x16x32_f16 per wave. Partial row sums are
// pure stores (deterministic).
// ---------------------------------------------------------------------------
__global__ void __launch_bounds__(128)
nt_simexp(const _Float16* __restrict__ zh, float* __restrict__ partial,
          int ntot, int mblocks, int colsPerChunk) {
  __shared__ __attribute__((aligned(32))) char smem[2 * BUF_BYTES];

  const int tid     = threadIdx.x;
  const int wave    = tid >> 5;
  const int lane    = tid & 31;
  const int lane16  = lane & 15;
  const int halfsel = lane >> 4;

  const int mblk  = blockIdx.x % mblocks;
  const int chunk = blockIdx.x / mblocks;
  const int Mbase = mblk * 64 + wave * 16;

  const unsigned lds_base = (unsigned)(uintptr_t)(&smem[0]);

  // Preload the 8 A fragments for this 16-row strip (K = 256 = 8 * 32).
  // ISA f16 A 16x32 layout: lanes 0-15 hold K {0..7,16..23}, lanes 16-31
  // hold K {8..15,24..31} of row lane%16.
  const _Float16* arow = zh + (size_t)(Mbase + lane16) * D_DIM + halfsel * 8;
  v16h afrag[8];
#pragma unroll
  for (int k = 0; k < 8; ++k) {
    union { v16h v; v8h h[2]; } u;
    u.h[0] = *(const v8h*)(arow + k * 32);
    u.h[1] = *(const v8h*)(arow + k * 32 + 16);
    afrag[k] = u.v;
  }

  float racc[8];
#pragma unroll
  for (int v = 0; v < 8; ++v) racc[v] = 0.f;

  const int nb0      = chunk * colsPerChunk;
  const int numTiles = colsPerChunk / TILE_N;   // 32

  // Cooperative async fill: 16 KB tile, 128 threads x 8 x 16B.
  auto fill = [&](int buf, int tile) {
    const int nb = nb0 + tile * TILE_N;
    const unsigned bufbase = lds_base + (unsigned)(buf * BUF_BYTES);
#pragma unroll
    for (int i = 0; i < 8; ++i) {
      const int cidx = tid + i * 128;        // 0..1023
      const int col  = cidx >> 5;            // 0..31
      const int q    = cidx & 31;            // 16B chunk within column
      async_copy_b128(bufbase + (unsigned)(col * COL_BYTES + q * 16),
                      zh + (size_t)(nb + col) * D_DIM + q * 8);
    }
  };

  fill(0, 0);
  int buf = 0;
  for (int t = 0; t < numTiles; ++t) {
    wait_async_all();          // this wave's fill of `buf` has landed in LDS
    __syncthreads();           // all waves' fills landed; prev reads of buf^1 done
    if (t + 1 < numTiles) fill(buf ^ 1, t + 1);   // overlaps with compute below

    const char* bufp = smem + buf * BUF_BYTES;
#pragma unroll
    for (int s = 0; s < 2; ++s) {          // two 16-column subtiles
      // B (K x 16) f16 layout: same K order as A for column lane%16,
      // lanes 0-15 -> K {0..7,16..23}, lanes 16-31 -> K {8..15,24..31}.
      const char* bcol = bufp + (s * 16 + lane16) * COL_BYTES + halfsel * 16;
      v8f c = {};
#pragma unroll
      for (int k = 0; k < 8; ++k) {
        union { v16h v; v8h h[2]; } ub;
        ub.h[0] = *(const v8h*)(bcol + k * 64);
        ub.h[1] = *(const v8h*)(bcol + k * 64 + 32);
        c = __builtin_amdgcn_wmma_f32_16x16x32_f16(
                false, afrag[k], false, ub.v, (short)0, c, false, false);
      }
      // exp(sim / T), T = 0.5; v_exp (TRANS) co-executes with XDL WMMA.
#pragma unroll
      for (int v = 0; v < 8; ++v) racc[v] += __expf(c[v] * 2.0f);
    }
    buf ^= 1;
  }

  // C/D layout: VGPR v holds row (v + halfsel*8), lane%16 = column.
#pragma unroll
  for (int v = 0; v < 8; ++v) {
    float s = racc[v];
    s += __shfl_xor(s, 1);  s += __shfl_xor(s, 2);
    s += __shfl_xor(s, 4);  s += __shfl_xor(s, 8);
    if (lane16 == 0)
      partial[(size_t)chunk * ntot + Mbase + halfsel * 8 + v] = s;
  }
}

// ---------------------------------------------------------------------------
// Kernel 4: deterministic scalar reduction.
// loss = (1/M^2) * sum_j [(M-1)*log(S_j) + log(max(S_j - 1, EPS))]
//        - (sum_j pos_total_j) / (T * M),   pos_total = 2 * sum_{j<N} pos_j
// ---------------------------------------------------------------------------
__global__ void nt_finalize(const float* __restrict__ partial,
                            const float* __restrict__ pos,
                            float* __restrict__ out,
                            int ntot, int n, int nchunks) {
  __shared__ float sh[256];
  const int tid = threadIdx.x;
  const float Mf = (float)ntot;

  float local = 0.f;
  for (int j = tid; j < ntot; j += 256) {
    float S = 0.f;
    for (int c = 0; c < nchunks; ++c) S += partial[(size_t)c * ntot + j];
    local += (Mf - 1.0f) * logf(S) + logf(fmaxf(S - 1.0f, EPS));
  }
  float plocal = 0.f;
  for (int j = tid; j < n; j += 256) plocal += pos[j];

  sh[tid] = local;  __syncthreads();
  for (int st = 128; st > 0; st >>= 1) {
    if (tid < st) sh[tid] += sh[tid + st];
    __syncthreads();
  }
  const float sumTerms = sh[0];
  __syncthreads();

  sh[tid] = plocal; __syncthreads();
  for (int st = 128; st > 0; st >>= 1) {
    if (tid < st) sh[tid] += sh[tid + st];
    __syncthreads();
  }
  if (tid == 0) {
    const float posTotal = 2.0f * sh[0];                 // both diagonals
    const float loss = sumTerms / (Mf * Mf) - posTotal * 2.0f / Mf; // /(T*M), T=0.5
    out[0] = loss;
  }
}

// ---------------------------------------------------------------------------
extern "C" void kernel_launch(void* const* d_in, const int* in_sizes, int n_in,
                              void* d_out, int out_size, void* d_ws, size_t ws_size,
                              hipStream_t stream) {
  const float* zi = (const float*)d_in[0];
  const float* zj = (const float*)d_in[1];
  float* out = (float*)d_out;

  const int n    = in_sizes[0] / D_DIM;   // 4096
  const int ntot = 2 * n;                 // 8192

  char* ws = (char*)d_ws;
  _Float16* zh      = (_Float16*)ws;                                   // ntot*256 f16
  float*    partial = (float*)(ws + (size_t)ntot * D_DIM * 2);         // NCHUNKS*ntot f32
  float*    pos     = (float*)((char*)partial + (size_t)NCHUNKS * ntot * 4); // n f32

  const int mblocks      = ntot / 64;        // 128 (4 waves x 16 rows per block)
  const int colsPerChunk = ntot / NCHUNKS;   // 1024

  nt_normalize<<<ntot / 4, 128, 0, stream>>>(zi, zj, zh, n);
  nt_positives<<<n / 4,    128, 0, stream>>>(zi, zj, pos, n);
  nt_simexp<<<mblocks * NCHUNKS, 128, 0, stream>>>(zh, partial, ntot, mblocks,
                                                   colsPerChunk);
  nt_finalize<<<1, 256, 0, stream>>>(partial, pos, out, ntot, n, NCHUNKS);
}
